// TransformerBlock_16166256902981
// MI455X (gfx1250) — compile-verified, matrix operand order checked by reading, not run on hardware
//
#include <hip/hip_runtime.h>
#include <hip/hip_bf16.h>

// ---------------- problem constants (match reference) ----------------
constexpr int B_  = 4;
constexpr int T_  = 1024;
constexpr int P_  = 1024;
constexpr int E_  = 1024;
constexpr int H_  = 16;
constexpr int D_  = 64;          // E/H
constexpr int FF_ = 4096;        // 4*E
constexpr int S_  = P_ + T_;     // 2048 keys
constexpr int M_  = B_ * T_;     // 4096 rows for all GEMMs

// ---------------- WMMA types ----------------
typedef _Float16 v16h __attribute__((ext_vector_type(16)));
typedef _Float16 v8h  __attribute__((ext_vector_type(8)));
typedef float    v8f  __attribute__((ext_vector_type(8)));

union Frag {
  v16h v;
  v8h  h[2];
  _Float16 e[16];
};

static __device__ inline v8f wmma16(const Frag& a, const Frag& b, v8f c) {
  // D = A(16x32 f16) * B(32x16 f16) + C(16x16 f32)
  return __builtin_amdgcn_wmma_f32_16x16x32_f16(
      /*neg_a=*/false, a.v, /*neg_b=*/false, b.v,
      /*c_mod=*/(short)0, c, /*reuse_a=*/false, /*reuse_b=*/false);
}

// ---------------- CDNA5 async memory->LDS (ASYNCcnt-tracked) ----------------
// VGLOBAL encoding: vdst = per-lane LDS byte address, vaddr = 64-bit global
// address, saddr = off. Low 32 bits of a generic LDS pointer are the LDS byte
// address (flat-aperture truncation rule, ISA 10.2).
static __device__ inline void async_b128(const _Float16* lds_dst,
                                         const _Float16* gsrc) {
  unsigned int l = (unsigned int)(size_t)lds_dst;
  unsigned long long g = (unsigned long long)(size_t)gsrc;
  asm volatile("global_load_async_to_lds_b128 %0, %1, off"
               :: "v"(l), "v"(g) : "memory");
}
static __device__ inline void async_wait() {
  asm volatile("s_wait_asynccnt 0x0" ::: "memory");
}

// ---------------- weight convert: W[K,N] f32 -> WT[N,K] f16 ----------------
__global__ __launch_bounds__(256) void wconv_k(const float* __restrict__ W,
                                               _Float16* __restrict__ WT,
                                               int K, int N) {
  int i = blockIdx.x * 256 + threadIdx.x;   // over K*N
  int k = i / N, n = i % N;
  WT[(size_t)n * K + k] = (_Float16)W[(size_t)k * N + n];
}

// ---------------- layernorm: x[row,E] f32 -> out f16 ----------------
__global__ __launch_bounds__(256) void ln_k(const float* __restrict__ x,
                                            const float* __restrict__ g,
                                            const float* __restrict__ bb,
                                            _Float16* __restrict__ out) {
  __shared__ float s1[256];
  __shared__ float s2[256];
  int row = blockIdx.x, tid = threadIdx.x;
  const float* xr = x + (size_t)row * E_;
  float v[4];
  float a0 = 0.f, a1 = 0.f;
#pragma unroll
  for (int i = 0; i < 4; ++i) {
    v[i] = xr[tid + 256 * i];
    a0 += v[i];
    a1 += v[i] * v[i];
  }
  s1[tid] = a0; s2[tid] = a1;
  __syncthreads();
  for (int s = 128; s > 0; s >>= 1) {
    if (tid < s) { s1[tid] += s1[tid + s]; s2[tid] += s2[tid + s]; }
    __syncthreads();
  }
  float mu  = s1[0] * (1.0f / E_);
  float var = s2[0] * (1.0f / E_) - mu * mu;
  float rs  = rsqrtf(var + 1e-5f);
#pragma unroll
  for (int i = 0; i < 4; ++i) {
    int c = tid + 256 * i;
    out[(size_t)row * E_ + c] = (_Float16)((v[i] - mu) * rs * g[c] + bb[c]);
  }
}

// ---------------- tiled WMMA GEMM ----------------
// C[M,N] = A[M,K](f16) * WT[N,K](f16)^T + bias, optional gelu, optional
// residual. Block tile 256(M) x 64(N), K-step 32 (= one WMMA K). 256 threads
// = 8 waves; wave w owns rows [32w,32w+32): 2 A-frags x 4 B-frags = 8 WMMAs
// per K-step. Tiles staged with CDNA5 async-to-LDS (no VGPR round-trip).
__global__ __launch_bounds__(256) void gemm_k(const _Float16* __restrict__ A,
                                              const _Float16* __restrict__ WT,
                                              const float* __restrict__ bias,
                                              const float* __restrict__ resid,
                                              float* __restrict__ outF,
                                              _Float16* __restrict__ outH,
                                              int N, int K, int dogelu) {
  __shared__ alignas(16) _Float16 la[256 * 32];  // A tile, row-major [256][32] (16KB)
  __shared__ alignas(16) _Float16 lb[64 * 32];   // WT tile, row-major [64][32] (4KB)
  const int tid  = threadIdx.x;
  const int wave = tid >> 5;
  const int lane = tid & 31;
  const int lg   = lane & 15;
  const int hi   = lane >> 4;
  const int koff = hi * 8;
  const int mb = blockIdx.y, nb = blockIdx.x;

  const _Float16* Ab = A  + (size_t)mb * 256 * K;
  const _Float16* Wb = WT + (size_t)nb * 64 * K;

  v8f acc[2][4];
#pragma unroll
  for (int fr = 0; fr < 2; ++fr)
#pragma unroll
    for (int nt = 0; nt < 4; ++nt) acc[fr][nt] = v8f{};

  for (int k0 = 0; k0 < K; k0 += 32) {
    // async A tile: 1024 x b128, 4 per thread
#pragma unroll
    for (int i = 0; i < 4; ++i) {
      int idx = tid + 256 * i;
      int r = idx >> 2, c = idx & 3;
      async_b128(la + r * 32 + c * 8, Ab + (size_t)r * K + k0 + c * 8);
    }
    // async WT tile: 256 x b128, 1 per thread
    {
      int r = tid >> 2, c = tid & 3;
      async_b128(lb + r * 32 + c * 8, Wb + (size_t)r * K + k0 + c * 8);
    }
    if (k0 + 32 < K) {  // CDNA5 global_prefetch_b8 on next tiles
      __builtin_prefetch(Ab + (size_t)tid * K + k0 + 32, 0, 1);
      __builtin_prefetch(Wb + (size_t)(tid >> 2) * K + k0 + 32, 0, 1);
    }
    async_wait();
    __syncthreads();

    Frag a[2];
#pragma unroll
    for (int fr = 0; fr < 2; ++fr) {
      const _Float16* ar = la + (wave * 32 + fr * 16 + lg) * 32;
      a[fr].h[0] = *(const v8h*)(ar + koff);
      a[fr].h[1] = *(const v8h*)(ar + koff + 16);
    }
#pragma unroll
    for (int nt = 0; nt < 4; ++nt) {
      Frag b;
      b.h[0] = *(const v8h*)(lb + (nt * 16 + lg) * 32 + koff);
      b.h[1] = *(const v8h*)(lb + (nt * 16 + lg) * 32 + koff + 16);
#pragma unroll
      for (int fr = 0; fr < 2; ++fr) acc[fr][nt] = wmma16(a[fr], b, acc[fr][nt]);
    }
    __syncthreads();
  }

  // epilogue: C element (VGPR j, lane) -> row j+8*hi, col lg within 16x16 tile
#pragma unroll
  for (int fr = 0; fr < 2; ++fr) {
#pragma unroll
    for (int nt = 0; nt < 4; ++nt) {
#pragma unroll
      for (int j = 0; j < 8; ++j) {
        int row = mb * 256 + wave * 32 + fr * 16 + j + 8 * hi;
        int col = nb * 64 + nt * 16 + lg;
        float v = acc[fr][nt][j] + bias[col];
        if (dogelu) v = 0.5f * v * (1.0f + erff(v * 0.70710678118f));
        if (resid) v += resid[(size_t)row * N + col];
        size_t o = (size_t)row * N + col;
        if (outF) outF[o] = v;
        if (outH) outH[o] = (_Float16)v;
      }
    }
  }
}

// ---------------- head-split scatters ----------------
// C[M,E] f32 -> qh[B,H,T,D] f16
__global__ __launch_bounds__(256) void scatter_q_k(const float* __restrict__ C,
                                                   _Float16* __restrict__ qh) {
  size_t i = (size_t)blockIdx.x * 256 + threadIdx.x;  // over B*T*E
  int b = (int)(i / ((size_t)T_ * E_));
  int r = (int)(i % ((size_t)T_ * E_));
  int t = r / E_, c = r % E_;
  int h = c / D_, d = c % D_;
  qh[(((size_t)b * H_ + h) * T_ + t) * D_ + d] = (_Float16)C[i];
}

// C[M,E] f32 -> cacheF[B,H,S,D] (at key index P+t) f32 + cacheH f16
__global__ __launch_bounds__(256) void scatter_kv_k(const float* __restrict__ C,
                                                    float* __restrict__ cacheF,
                                                    _Float16* __restrict__ cacheH) {
  size_t i = (size_t)blockIdx.x * 256 + threadIdx.x;  // over B*T*E
  int b = (int)(i / ((size_t)T_ * E_));
  int r = (int)(i % ((size_t)T_ * E_));
  int t = r / E_, c = r % E_;
  int h = c / D_, d = c % D_;
  size_t o = (((size_t)b * H_ + h) * S_ + (P_ + t)) * D_ + d;
  float v = C[i];
  cacheF[o] = v;
  cacheH[o] = (_Float16)v;
}

// pk[B,H,P,D] f32 -> cacheF[B,H,S,D] prefix + cacheH f16
__global__ __launch_bounds__(256) void prefix_kv_k(const float* __restrict__ pk,
                                                   float* __restrict__ cacheF,
                                                   _Float16* __restrict__ cacheH) {
  size_t i = (size_t)blockIdx.x * 256 + threadIdx.x;  // over B*H*P*D
  size_t bh = i / ((size_t)P_ * D_);
  int r = (int)(i % ((size_t)P_ * D_));
  int p = r / D_, d = r % D_;
  size_t o = (bh * S_ + p) * D_ + d;
  float v = pk[i];
  cacheF[o] = v;
  cacheH[o] = (_Float16)v;
}

// ---------------- flash attention: one wave per (b,h,16-query tile) ----------------
__global__ __launch_bounds__(32) void attn_k(const _Float16* __restrict__ qh, // [B,H,T,D]
                                             const _Float16* __restrict__ kh, // [B,H,S,D]
                                             const _Float16* __restrict__ vh, // [B,H,S,D]
                                             _Float16* __restrict__ oh) {     // [B,T,E]
  __shared__ alignas(16) _Float16 lds_p[16 * 32];  // P tile (16 q x 32 keys)
  __shared__ alignas(16) _Float16 lds_v[32 * 64];  // V chunk (32 keys x 64 d)

  const int tile = blockIdx.x;                 // B*H*(T/16) tiles
  const int qt   = tile % (T_ / 16);
  const int h    = (tile / (T_ / 16)) % H_;
  const int b    = tile / ((T_ / 16) * H_);
  const int lane = threadIdx.x;
  const int lg   = lane & 15;
  const int hi   = lane >> 4;
  const int koff = hi * 8;

  const _Float16* qb = qh + (((size_t)b * H_ + h) * T_ + (size_t)qt * 16) * D_;
  const _Float16* kb = kh + ((size_t)b * H_ + h) * S_ * D_;
  const _Float16* vb = vh + ((size_t)b * H_ + h) * S_ * D_;

  // Q fragments (16x32 each; D=64 -> two)
  Frag qa0, qa1;
  {
    const _Float16* r = qb + (size_t)lg * D_;
    qa0.h[0] = *(const v8h*)(r + koff);
    qa0.h[1] = *(const v8h*)(r + koff + 16);
    qa1.h[0] = *(const v8h*)(r + 32 + koff);
    qa1.h[1] = *(const v8h*)(r + 32 + koff + 16);
  }

  v8f acc[4] = {v8f{}, v8f{}, v8f{}, v8f{}};
  float m[8], l[8];
#pragma unroll
  for (int j = 0; j < 8; ++j) { m[j] = -1e30f; l[j] = 0.f; }

  const int kmax = P_ + qt * 16 + 15;  // last key any row of this tile may see
  for (int s0 = 0; s0 < S_ && s0 <= kmax; s0 += 32) {
    __syncthreads();  // protect lds reuse across iterations

    // async-stage V chunk [32 keys][64 d] row-major; overlaps QK^T + softmax
#pragma unroll
    for (int i = 0; i < 8; ++i) {
      int idx  = lane + 32 * i;
      int krel = idx >> 3, c8 = idx & 7;
      async_b128(lds_v + krel * 64 + c8 * 8,
                 vb + (size_t)(s0 + krel) * D_ + c8 * 8);
    }

    // scores for two 16-key subtiles: S = Q * K^T (K frag read contiguous)
    v8f sc[2];
#pragma unroll
    for (int st = 0; st < 2; ++st) {
      const _Float16* kr = kb + (size_t)(s0 + st * 16 + lg) * D_;
      Frag b0, b1;
      b0.h[0] = *(const v8h*)(kr + koff);
      b0.h[1] = *(const v8h*)(kr + koff + 16);
      b1.h[0] = *(const v8h*)(kr + 32 + koff);
      b1.h[1] = *(const v8h*)(kr + 32 + koff + 16);
      v8f c = {};
      c = wmma16(qa0, b0, c);
      c = wmma16(qa1, b1, c);
      sc[st] = c;
    }

    // online softmax, C-fragment rows: row = j + 8*hi
#pragma unroll
    for (int j = 0; j < 8; ++j) {
      int row = qt * 16 + j + 8 * hi;
      int lim = P_ + row;
      float s0v = (s0 + lg      <= lim) ? sc[0][j] * 0.125f : -1e30f;
      float s1v = (s0 + 16 + lg <= lim) ? sc[1][j] * 0.125f : -1e30f;
      float bm = fmaxf(s0v, s1v);
      bm = fmaxf(bm, __shfl_xor(bm, 1));
      bm = fmaxf(bm, __shfl_xor(bm, 2));
      bm = fmaxf(bm, __shfl_xor(bm, 4));
      bm = fmaxf(bm, __shfl_xor(bm, 8));
      float mn = fmaxf(m[j], bm);
      float f  = __expf(m[j] - mn);
      float p0 = __expf(s0v - mn);
      float p1 = __expf(s1v - mn);
      float rs = p0 + p1;
      rs += __shfl_xor(rs, 1);
      rs += __shfl_xor(rs, 2);
      rs += __shfl_xor(rs, 4);
      rs += __shfl_xor(rs, 8);
      l[j] = l[j] * f + rs;
      m[j] = mn;
      acc[0][j] *= f; acc[1][j] *= f; acc[2][j] *= f; acc[3][j] *= f;
      lds_p[(j + 8 * hi) * 32 + lg]      = (_Float16)p0;
      lds_p[(j + 8 * hi) * 32 + 16 + lg] = (_Float16)p1;
    }
    async_wait();     // V chunk landed in LDS
    __syncthreads();

    // P fragment (A layout) and O += P * V
    Frag pf;
    pf.h[0] = *(const v8h*)(lds_p + lg * 32 + koff);
    pf.h[1] = *(const v8h*)(lds_p + lg * 32 + koff + 16);
#pragma unroll
    for (int nt = 0; nt < 4; ++nt) {
      Frag vf;  // B layout: lane column n = nt*16+lg, K = keys
#pragma unroll
      for (int i = 0; i < 8; ++i) {
        vf.e[i]     = lds_v[(koff + i)      * 64 + nt * 16 + lg];
        vf.e[i + 8] = lds_v[(koff + 16 + i) * 64 + nt * 16 + lg];
      }
      acc[nt] = wmma16(pf, vf, acc[nt]);
    }
  }

  // write merged-head output f16 [B,T,E]
#pragma unroll
  for (int nt = 0; nt < 4; ++nt) {
#pragma unroll
    for (int j = 0; j < 8; ++j) {
      int t   = qt * 16 + j + 8 * hi;
      int col = h * D_ + nt * 16 + lg;
      oh[((size_t)b * T_ + t) * E_ + col] = (_Float16)(acc[nt][j] / l[j]);
    }
  }
}

// ---------------- launch ----------------
extern "C" void kernel_launch(void* const* d_in, const int* in_sizes, int n_in,
                              void* d_out, int out_size, void* d_ws, size_t ws_size,
                              hipStream_t stream) {
  (void)in_sizes; (void)n_in; (void)out_size; (void)ws_size;
  const float* x    = (const float*)d_in[0];
  const float* pk   = (const float*)d_in[1];
  const float* pv   = (const float*)d_in[2];
  // d_in[3] = mask (computed analytically in-kernel)
  const float* wq   = (const float*)d_in[4];
  const float* bq   = (const float*)d_in[5];
  const float* wk   = (const float*)d_in[6];
  const float* bk   = (const float*)d_in[7];
  const float* wv   = (const float*)d_in[8];
  const float* bv   = (const float*)d_in[9];
  const float* wo   = (const float*)d_in[10];
  const float* bo   = (const float*)d_in[11];
  const float* g1   = (const float*)d_in[12];
  const float* b1   = (const float*)d_in[13];
  const float* g2   = (const float*)d_in[14];
  const float* b2   = (const float*)d_in[15];
  const float* wff1 = (const float*)d_in[16];
  const float* bff1 = (const float*)d_in[17];
  const float* wff2 = (const float*)d_in[18];
  const float* bff2 = (const float*)d_in[19];

  // outputs: x [B,T,E] | k [B,H,S,D] | v [B,H,S,D]
  float* out_x = (float*)d_out;
  float* out_k = out_x + (size_t)B_ * T_ * E_;
  float* out_v = out_k + (size_t)B_ * H_ * S_ * D_;

  // workspace carve-up
  char* w = (char*)d_ws;
  size_t off = 0;
  auto carve = [&](size_t bytes) {
    void* p = w + off;
    off += (bytes + 255) & ~(size_t)255;
    return p;
  };
  _Float16* wqT   = (_Float16*)carve((size_t)E_ * E_ * 2);
  _Float16* wkT   = (_Float16*)carve((size_t)E_ * E_ * 2);
  _Float16* wvT   = (_Float16*)carve((size_t)E_ * E_ * 2);
  _Float16* woT   = (_Float16*)carve((size_t)E_ * E_ * 2);
  _Float16* wff1T = (_Float16*)carve((size_t)E_ * FF_ * 2);
  _Float16* wff2T = (_Float16*)carve((size_t)FF_ * E_ * 2);
  _Float16* xn    = (_Float16*)carve((size_t)M_ * E_ * 2);
  _Float16* q_h   = (_Float16*)carve((size_t)B_ * H_ * T_ * D_ * 2);
  _Float16* k_h   = (_Float16*)carve((size_t)B_ * H_ * S_ * D_ * 2);
  _Float16* v_h   = (_Float16*)carve((size_t)B_ * H_ * S_ * D_ * 2);
  _Float16* o_h   = (_Float16*)carve((size_t)M_ * E_ * 2);
  float*    x1    = (float*)   carve((size_t)M_ * E_ * 4);
  _Float16* h2    = (_Float16*)carve((size_t)M_ * E_ * 2);
  _Float16* ff_h  = (_Float16*)carve((size_t)M_ * FF_ * 2);
  float*    cbuf  = (float*)   carve((size_t)M_ * E_ * 4);

  // 1) weight transposes -> f16 [N,K]
  wconv_k<<<(E_ * E_) / 256, 256, 0, stream>>>(wq, wqT, E_, E_);
  wconv_k<<<(E_ * E_) / 256, 256, 0, stream>>>(wk, wkT, E_, E_);
  wconv_k<<<(E_ * E_) / 256, 256, 0, stream>>>(wv, wvT, E_, E_);
  wconv_k<<<(E_ * E_) / 256, 256, 0, stream>>>(wo, woT, E_, E_);
  wconv_k<<<(E_ * FF_) / 256, 256, 0, stream>>>(wff1, wff1T, E_, FF_);
  wconv_k<<<(FF_ * E_) / 256, 256, 0, stream>>>(wff2, wff2T, FF_, E_);

  // 2) LN1
  ln_k<<<M_, 256, 0, stream>>>(x, g1, b1, xn);

  dim3 gE(E_ / 64, M_ / 256);   // N=1024 GEMMs
  dim3 gF(FF_ / 64, M_ / 256);  // N=4096 GEMM
  const int nElem = (M_ * E_) / 256;

  // 3) Q/K/V projections + head split / cache fill
  gemm_k<<<gE, 256, 0, stream>>>(xn, wqT, bq, nullptr, cbuf, nullptr, E_, E_, 0);
  scatter_q_k<<<nElem, 256, 0, stream>>>(cbuf, q_h);
  gemm_k<<<gE, 256, 0, stream>>>(xn, wkT, bk, nullptr, cbuf, nullptr, E_, E_, 0);
  scatter_kv_k<<<nElem, 256, 0, stream>>>(cbuf, out_k, k_h);
  gemm_k<<<gE, 256, 0, stream>>>(xn, wvT, bv, nullptr, cbuf, nullptr, E_, E_, 0);
  scatter_kv_k<<<nElem, 256, 0, stream>>>(cbuf, out_v, v_h);
  const int nPref = (B_ * H_ * P_ * D_) / 256;
  prefix_kv_k<<<nPref, 256, 0, stream>>>(pk, out_k, k_h);
  prefix_kv_k<<<nPref, 256, 0, stream>>>(pv, out_v, v_h);

  // 4) attention: one wave per 16-query tile
  attn_k<<<B_ * H_ * (T_ / 16), 32, 0, stream>>>(q_h, k_h, v_h, o_h);

  // 5) output projection + residual(x) -> x1 (f32)
  gemm_k<<<gE, 256, 0, stream>>>(o_h, woT, bo, x, x1, nullptr, E_, E_, 0);

  // 6) LN2
  ln_k<<<M_, 256, 0, stream>>>(x1, g2, b2, h2);

  // 7) FFN: gelu(h2 @ wff1 + b) -> ff_h (f16); ff_h @ wff2 + b + x1 -> out_x
  gemm_k<<<gF, 256, 0, stream>>>(h2, wff1T, bff1, nullptr, nullptr, ff_h, FF_, E_, 1);
  gemm_k<<<gE, 256, 0, stream>>>(ff_h, wff2T, bff2, x1, out_x, nullptr, E_, FF_, 0);
}
